// SimplifiedMoEBlock_46420006535172
// MI455X (gfx1250) — matrix operand order repaired and checked by reading, not compiled, not run on hardware
//
#include <hip/hip_runtime.h>

#define T_TOK 2048      // B*S tokens
#define H_DIM 2048
#define E_NUM 16
#define I_DIM 1408
#define BM    32            // token rows per block (two 16-row WMMA tiles sharing B)
#define IPAD  (I_DIM + 8)   // LDS row pad: 708 dwords % 64 banks = 4 -> conflict-free b128 reads

typedef __attribute__((ext_vector_type(16))) __bf16 v16bf;
typedef __attribute__((ext_vector_type(8)))  __bf16 v8bf;
typedef __attribute__((ext_vector_type(4)))  __bf16 v4bf;
typedef __attribute__((ext_vector_type(8)))  float  v8f;
typedef __attribute__((ext_vector_type(4)))  float  v4f;

static __device__ __forceinline__ v16bf cat8(v8bf lo, v8bf hi) {
  return __builtin_shufflevector(lo, hi, 0,1,2,3,4,5,6,7,8,9,10,11,12,13,14,15);
}
static __device__ __forceinline__ v8f wmma_bf16(v16bf a, v16bf b, v8f c) {
  // D = A(16x32 bf16) * B(32x16 bf16) + C(16x16 f32)
  return __builtin_amdgcn_wmma_f32_16x16x32_bf16(false, a, false, b, (short)0, c, false, false);
}

// A fragment (16x32, row-major bf16): lane r=lane&15 -> row, kbA=(lane>>4)*16,
// elements 0..15 = K[kbA..kbA+15]
static __device__ __forceinline__ v16bf load_a(const __bf16* row_k0, int kbA) {
  v8bf lo = *(const v8bf*)(row_k0 + kbA);
  v8bf hi = *(const v8bf*)(row_k0 + kbA + 8);
  return cat8(lo, hi);
}
// B fragment (32x16) from an fp32 weight row (column of B = contiguous K of W):
// lane c=lane&15 -> column, kbB=(lane>>4)*16, elements 0..15 = K[kbB..kbB+15].
// Loads 64B fp32 (4x b128) and converts to bf16 in registers -> no pre-conversion pass.
static __device__ __forceinline__ v16bf load_b_f32(const float* col_k0_kb) {
  v4f a0 = *(const v4f*)(col_k0_kb);
  v4f a1 = *(const v4f*)(col_k0_kb + 4);
  v4f a2 = *(const v4f*)(col_k0_kb + 8);
  v4f a3 = *(const v4f*)(col_k0_kb + 12);
  v16bf b;
#pragma unroll
  for (int j = 0; j < 4; ++j) {
    b[j]      = (__bf16)a0[j];
    b[4 + j]  = (__bf16)a1[j];
    b[8 + j]  = (__bf16)a2[j];
    b[12 + j] = (__bf16)a3[j];
  }
  return b;
}

// ---------------- gate: logits -> top-2 -> scatter routing lists ----------------
__global__ __launch_bounds__(256) void gate_topk_kernel(
    const float* __restrict__ x, const float* __restrict__ gw,
    int* __restrict__ counts, int* __restrict__ toks, float* __restrict__ wts) {
  const int t   = blockIdx.x;
  const int tid = threadIdx.x;
  const float* xp = x + (size_t)t * H_DIM;

  float acc[E_NUM];
#pragma unroll
  for (int e = 0; e < E_NUM; ++e) acc[e] = 0.0f;

#pragma unroll
  for (int i = 0; i < H_DIM / 256; ++i) {
    const int h = tid + i * 256;
    const float xv = xp[h];
#pragma unroll
    for (int e = 0; e < E_NUM; ++e) acc[e] += xv * gw[e * H_DIM + h];
  }
#pragma unroll
  for (int e = 0; e < E_NUM; ++e)
    for (int m = 16; m >= 1; m >>= 1) acc[e] += __shfl_xor(acc[e], m, 32);

  __shared__ float part[8][E_NUM];
  const int wave = tid >> 5, lane = tid & 31;
  if (lane == 0) {
#pragma unroll
    for (int e = 0; e < E_NUM; ++e) part[wave][e] = acc[e];
  }
  __syncthreads();
  if (tid == 0) {
    float lg[E_NUM];
#pragma unroll
    for (int e = 0; e < E_NUM; ++e) {
      float s = 0.0f;
#pragma unroll
      for (int w = 0; w < 8; ++w) s += part[w][e];
      lg[e] = s;
    }
    int i0 = 0;
#pragma unroll
    for (int e = 1; e < E_NUM; ++e) if (lg[e] > lg[i0]) i0 = e;
    int i1 = (i0 == 0) ? 1 : 0;
#pragma unroll
    for (int e = 0; e < E_NUM; ++e) if (e != i0 && lg[e] > lg[i1]) i1 = e;
    // softmax over top-2 == renormalized full softmax restricted to top-2
    const float w0 = 1.0f / (1.0f + __expf(lg[i1] - lg[i0]));
    const float w1 = 1.0f - w0;
    int p0 = atomicAdd(&counts[i0], 1);
    toks[i0 * T_TOK + p0] = t; wts[i0 * T_TOK + p0] = w0;
    int p1 = atomicAdd(&counts[i1], 1);
    toks[i1 * T_TOK + p1] = t; wts[i1 * T_TOK + p1] = w1;
  }
}

// ---------------- fp32 -> bf16 conversion for x only (4-wide) ----------------
__global__ __launch_bounds__(256) void cvt_f32_to_bf16(
    const float* __restrict__ in, __bf16* __restrict__ out, int n4) {
  const int i = blockIdx.x * 256 + threadIdx.x;
  if (i >= n4) return;
  v4f v = *(const v4f*)(in + 4 * (size_t)i);
  v4bf o;
  o[0] = (__bf16)v[0]; o[1] = (__bf16)v[1]; o[2] = (__bf16)v[2]; o[3] = (__bf16)v[3];
  *(v4bf*)(out + 4 * (size_t)i) = o;
}

// ---------------- fused per-expert SwiGLU FFN (bf16 WMMA, fp32 weights in-flight) ----------------
__global__ __launch_bounds__(256) void moe_ffn_kernel(
    const __bf16* __restrict__ xb, const float* __restrict__ wg,
    const float* __restrict__ wu, const float* __restrict__ wd,
    const int* __restrict__ counts, const int* __restrict__ toks,
    const float* __restrict__ wts, float* __restrict__ out) {
  const int e    = blockIdx.y;
  const int tile = blockIdx.x;
  const int cnt  = counts[e];
  const int row0 = tile * BM;
  if (row0 >= cnt) return;
  const int rows = (cnt - row0 < BM) ? (cnt - row0) : BM;

  __shared__ int    stok[BM];
  __shared__ float  swt[BM];
  __shared__ __bf16 sact[BM * IPAD];   // 32 tokens x I activation tile (~90KB of 320KB WGP LDS)

  const int tid = threadIdx.x;
  if (tid < BM) {
    if (tid < rows) {
      stok[tid] = toks[e * T_TOK + row0 + tid];
      swt[tid]  = wts[e * T_TOK + row0 + tid];
    } else {
      stok[tid] = 0; swt[tid] = 0.0f;
    }
  }
  __syncthreads();

  const int lane  = tid & 31;
  const int wave  = tid >> 5;
  const int r     = lane & 15;          // A row (within 16-row tile) / B col / C col
  const int kbA   = (lane >> 4) << 4;   // A K-base: 0 or 16
  const int kbB   = (lane >> 4) << 4;   // B K-base: 0 or 16
  const int mbase = (lane >> 4) << 3;   // C row base: 0 or 8

  // two 16-row A tiles; pad rows clamp to a real token, masked at the output store
  const int rr0 = (r < rows) ? r : (rows - 1);
  const int r16 = r + 16;
  const int rr1 = (r16 < rows) ? r16 : (rows - 1);
  const __bf16* xrow0 = xb + (size_t)stok[rr0] * H_DIM;
  const __bf16* xrow1 = xb + (size_t)stok[rr1] * H_DIM;

  const float* wge = wg + (size_t)e * I_DIM * H_DIM;
  const float* wue = wu + (size_t)e * I_DIM * H_DIM;

  // -------- phase 1: act[32 x I] = silu(x Wg^T) * (x Wu^T), K = H --------
  for (int it = wave; it < I_DIM / 16; it += 8) {
    const int i0 = it * 16;
    v8f g0 = {}, u0 = {}, g1 = {}, u1 = {};
    const float* bg = wge + (size_t)(i0 + r) * H_DIM + kbB;
    const float* bu = wue + (size_t)(i0 + r) * H_DIM + kbB;
#pragma unroll 2
    for (int k0 = 0; k0 < H_DIM; k0 += 32) {
      v16bf a0 = load_a(xrow0 + k0, kbA);
      v16bf a1 = load_a(xrow1 + k0, kbA);
      v16bf bgv = load_b_f32(bg + k0);     // fp32 -> bf16 in registers
      g0 = wmma_bf16(a0, bgv, g0);
      g1 = wmma_bf16(a1, bgv, g1);
      v16bf buv = load_b_f32(bu + k0);
      u0 = wmma_bf16(a0, buv, u0);
      u1 = wmma_bf16(a1, buv, u1);
    }
#pragma unroll
    for (int j = 0; j < 8; ++j) {
      const float gv0 = g0[j];
      const float gv1 = g1[j];
      const float s0  = gv0 / (1.0f + __expf(-gv0));   // silu
      const float s1  = gv1 / (1.0f + __expf(-gv1));
      sact[(mbase + j) * IPAD + i0 + r]        = (__bf16)(s0 * u0[j]);
      sact[(16 + mbase + j) * IPAD + i0 + r]   = (__bf16)(s1 * u1[j]);
    }
  }
  __syncthreads();

  // -------- phase 2: y[32 x H] = act @ Wd^T, K = I; scale + atomic combine --------
  const float* wde = wd + (size_t)e * (size_t)H_DIM * I_DIM;
  for (int ht = wave; ht < H_DIM / 16; ht += 8) {
    const int h0 = ht * 16;
    v8f c0 = {}, c1 = {};
    const float* bp  = wde + (size_t)(h0 + r) * I_DIM + kbB;
    const __bf16* a0p = sact + r * IPAD;
    const __bf16* a1p = sact + (r + 16) * IPAD;
#pragma unroll 2
    for (int k0 = 0; k0 < I_DIM; k0 += 32) {
      v16bf a0 = load_a(a0p + k0, kbA);    // ds_load_b128 x2, conflict-free via IPAD
      v16bf a1 = load_a(a1p + k0, kbA);
      v16bf b  = load_b_f32(bp + k0);      // fp32 -> bf16 in registers
      c0 = wmma_bf16(a0, b, c0);
      c1 = wmma_bf16(a1, b, c1);
    }
#pragma unroll
    for (int j = 0; j < 8; ++j) {
      const int m0 = mbase + j;
      const int m1 = m0 + 16;
      if (m0 < rows)
        unsafeAtomicAdd(out + (size_t)stok[m0] * H_DIM + h0 + r, c0[j] * swt[m0]);
      if (m1 < rows)
        unsafeAtomicAdd(out + (size_t)stok[m1] * H_DIM + h0 + r, c1[j] * swt[m1]);
    }
  }
}

extern "C" void kernel_launch(void* const* d_in, const int* in_sizes, int n_in,
                              void* d_out, int out_size, void* d_ws, size_t ws_size,
                              hipStream_t stream) {
  (void)in_sizes; (void)n_in; (void)ws_size;
  const float* x  = (const float*)d_in[0];   // (B,S,H)
  const float* gw = (const float*)d_in[1];   // (E,H)
  const float* wg = (const float*)d_in[2];   // (E,I,H)
  const float* wu = (const float*)d_in[3];   // (E,I,H)
  const float* wd = (const float*)d_in[4];   // (E,H,I)
  float* out = (float*)d_out;

  char* ws = (char*)d_ws;
  size_t off = 0;
  auto carve = [&](size_t bytes) -> void* {
    off = (off + 255) & ~(size_t)255;
    void* p = ws + off;
    off += bytes;
    return p;
  };
  int*    counts = (int*)carve(E_NUM * sizeof(int));
  int*    toks   = (int*)carve((size_t)E_NUM * T_TOK * sizeof(int));
  float*  wts    = (float*)carve((size_t)E_NUM * T_TOK * sizeof(float));
  __bf16* xb     = (__bf16*)carve((size_t)T_TOK * H_DIM * sizeof(__bf16));

  hipMemsetAsync(counts, 0, E_NUM * sizeof(int), stream);
  hipMemsetAsync(d_out, 0, (size_t)out_size * sizeof(float), stream);

  gate_topk_kernel<<<T_TOK, 256, 0, stream>>>(x, gw, counts, toks, wts);

  const int n4x = T_TOK * H_DIM / 4;
  cvt_f32_to_bf16<<<(n4x + 255) / 256, 256, 0, stream>>>(x, xb, n4x);

  dim3 grid(T_TOK / BM, E_NUM);
  moe_ffn_kernel<<<grid, 256, 0, stream>>>(xb, wg, wu, wd, counts, toks, wts, out);
}
